// TransitionUp_29480655520250
// MI455X (gfx1250) — compile-verified, matrix operand order checked by reading, not run on hardware
//
#include <hip/hip_runtime.h>
#include <hip/hip_bf16.h>

typedef __attribute__((ext_vector_type(16))) __bf16          v16bf;
typedef __attribute__((ext_vector_type(16))) unsigned short  v16u;
typedef __attribute__((ext_vector_type(8)))  float           v8f;

#define BATCH 4
#define NP1   8192
#define NP2   2048
#define CIN   512
#define COUT  256

#define USE_ASYNC_LDS 1

// ---------------- helpers ----------------
__device__ __forceinline__ unsigned short f32_to_bf16_rne(float f) {
    unsigned int u = __float_as_uint(f);
    unsigned int r = u + 0x7FFFu + ((u >> 16) & 1u);
    return (unsigned short)(r >> 16);
}

__device__ __forceinline__ void wait_async0() {
#if __has_builtin(__builtin_amdgcn_s_wait_asynccnt)
    __builtin_amdgcn_s_wait_asynccnt(0);
#else
    asm volatile("s_wait_asynccnt 0x0" ::: "memory");
#endif
}

// async copy of 16 bytes global -> LDS (CDNA5 GLOBAL_LOAD_ASYNC_TO_LDS_B128)
__device__ __forceinline__ void async_b128(unsigned int ldsOff, const unsigned short* g) {
#if USE_ASYNC_LDS
    asm volatile("global_load_async_to_lds_b128 %0, %1, off"
                 :: "v"(ldsOff), "v"((unsigned long long)(uintptr_t)g)
                 : "memory");
#endif
}

// ---------------- one-shot conversions ----------------
// float -> bf16, 8 elements per thread (vectorized b128 in / b128 out)
__global__ __launch_bounds__(256) void cvt_bf16_kernel(
    const float* __restrict__ src, unsigned short* __restrict__ dst, int n8)
{
    int i = blockIdx.x * 256 + threadIdx.x;
    if (i >= n8) return;
    const float4* s4 = (const float4*)src;
    float4 a = s4[2 * i], b = s4[2 * i + 1];
    uint4 o;
    o.x = (unsigned)f32_to_bf16_rne(a.x) | ((unsigned)f32_to_bf16_rne(a.y) << 16);
    o.y = (unsigned)f32_to_bf16_rne(a.z) | ((unsigned)f32_to_bf16_rne(a.w) << 16);
    o.z = (unsigned)f32_to_bf16_rne(b.x) | ((unsigned)f32_to_bf16_rne(b.y) << 16);
    o.w = (unsigned)f32_to_bf16_rne(b.z) | ((unsigned)f32_to_bf16_rne(b.w) << 16);
    ((uint4*)dst)[i] = o;
}

// W[K,256] f32  ->  Wt[256,K] bf16   (N hardcoded = COUT = 256)
__global__ __launch_bounds__(256) void cvt_w_transpose_kernel(
    const float* __restrict__ W, unsigned short* __restrict__ Wt, int K)
{
    int e = blockIdx.x * 256 + threadIdx.x;   // e < K*256
    int k = e >> 8, n = e & 255;
    Wt[(size_t)n * K + k] = f32_to_bf16_rne(W[e]);
}

// ---------------- GEMM: Y[M,N] = A[M,K] @ Wt[N,K]^T + bias[N] ----------------
// bf16 WMMA, f32 accum; async double-buffered LDS staging
#define BM 64
#define BN 128
#define KB 32
#define ASTR 40   // padded LDS row stride (ushorts): 80B rows, 16B-aligned chunks

__global__ __launch_bounds__(256) void gemm_bias_bf16(
    const unsigned short* __restrict__ A,    // [M,K] bf16
    const unsigned short* __restrict__ Wt,   // [N,K] bf16
    const float* __restrict__ bias, float* __restrict__ Y,
    int M, int N, int K)
{
    __shared__ unsigned short As[2][BM * ASTR];
    __shared__ unsigned short Bs[2][BN * ASTR];

    const int tid  = threadIdx.x;
    const int lane = tid & 31;
    const int wave = tid >> 5;      // 0..7
    const int wm   = wave >> 2;     // 0..1 : 32-row slab
    const int wn   = wave & 3;      // 0..3 : 32-col slab
    const int half = lane >> 4;     // 0/1
    const int mr   = lane & 15;

    const int mBlock = blockIdx.y * BM;
    const int nBlock = blockIdx.x * BN;

    // staging: thread -> (row sr, 8-elem chunk sc); one b128 for A, two for B
    const int sr = tid >> 2;          // 0..63
    const int sc = (tid & 3) * 8;     // 0,8,16,24

    const unsigned short* gA  = A  + (size_t)(mBlock + sr) * K + sc;
    const unsigned short* gB0 = Wt + (size_t)(nBlock + sr) * K + sc;
    const unsigned short* gB1 = Wt + (size_t)(nBlock + 64 + sr) * K + sc;

    unsigned int aOff[2], b0Off[2], b1Off[2];
#pragma unroll
    for (int b = 0; b < 2; ++b) {
        aOff[b]  = (unsigned int)(uintptr_t)&As[b][sr * ASTR + sc];
        b0Off[b] = (unsigned int)(uintptr_t)&Bs[b][sr * ASTR + sc];
        b1Off[b] = (unsigned int)(uintptr_t)&Bs[b][(64 + sr) * ASTR + sc];
    }

    v8f acc[2][2];
#pragma unroll
    for (int ti = 0; ti < 2; ++ti)
#pragma unroll
        for (int tj = 0; tj < 2; ++tj) {
            float bv = bias[nBlock + wn * 32 + tj * 16 + mr]; // per-lane N fixed
            v8f t;
#pragma unroll
            for (int v = 0; v < 8; ++v) t[v] = bv;
            acc[ti][tj] = t;
        }

    const int nK = K / KB;

#if USE_ASYNC_LDS
    async_b128(aOff[0], gA);
    async_b128(b0Off[0], gB0);
    async_b128(b1Off[0], gB1);
    wait_async0();
#else
    *(uint4*)((char*)&As[0][0] + (aOff[0]  - (unsigned int)(uintptr_t)&As[0][0])) = *(const uint4*)gA;
    *(uint4*)((char*)&Bs[0][0] + (b0Off[0] - (unsigned int)(uintptr_t)&Bs[0][0])) = *(const uint4*)gB0;
    *(uint4*)((char*)&Bs[0][0] + (b1Off[0] - (unsigned int)(uintptr_t)&Bs[0][0])) = *(const uint4*)gB1;
#endif
    __syncthreads();

    for (int ks = 0; ks < nK; ++ks) {
        const int cur = ks & 1;
        if (ks + 1 < nK) {
            const int kofs = (ks + 1) * KB;
#if USE_ASYNC_LDS
            async_b128(aOff[cur ^ 1],  gA  + kofs);
            async_b128(b0Off[cur ^ 1], gB0 + kofs);
            async_b128(b1Off[cur ^ 1], gB1 + kofs);
#else
            ((uint4*)&As[cur ^ 1][sr * ASTR + sc])[0]        = *(const uint4*)(gA  + kofs);
            ((uint4*)&Bs[cur ^ 1][sr * ASTR + sc])[0]        = *(const uint4*)(gB0 + kofs);
            ((uint4*)&Bs[cur ^ 1][(64 + sr) * ASTR + sc])[0] = *(const uint4*)(gB1 + kofs);
#endif
        }

        // fragments per ISA 16-bit 16x32 layout:
        // VGPR v holds K = ((v&4)?16:0) + (v&3)*2 + 8*(lane>>4), pair {k,k+1}
        v16u afrag[2], bfrag[2];
#pragma unroll
        for (int ti = 0; ti < 2; ++ti) {
            int m = wm * 32 + ti * 16 + mr;
#pragma unroll
            for (int v = 0; v < 8; ++v) {
                int k = ((v & 4) ? 16 : 0) + (v & 3) * 2 + 8 * half;
                afrag[ti][2 * v]     = As[cur][m * ASTR + k];
                afrag[ti][2 * v + 1] = As[cur][m * ASTR + k + 1];
            }
        }
#pragma unroll
        for (int tj = 0; tj < 2; ++tj) {
            int n = wn * 32 + tj * 16 + mr;
#pragma unroll
            for (int v = 0; v < 8; ++v) {
                int k = ((v & 4) ? 16 : 0) + (v & 3) * 2 + 8 * half;
                bfrag[tj][2 * v]     = Bs[cur][n * ASTR + k];
                bfrag[tj][2 * v + 1] = Bs[cur][n * ASTR + k + 1];
            }
        }
#pragma unroll
        for (int ti = 0; ti < 2; ++ti)
#pragma unroll
            for (int tj = 0; tj < 2; ++tj)
                acc[ti][tj] = __builtin_amdgcn_wmma_f32_16x16x32_bf16(
                    false, __builtin_bit_cast(v16bf, afrag[ti]),
                    false, __builtin_bit_cast(v16bf, bfrag[tj]),
                    (short)0, acc[ti][tj], false, false);

        if (ks + 1 < nK) {
#if USE_ASYNC_LDS
            wait_async0();
#endif
            __syncthreads();
        }
    }

    // store: C/D layout — VGPR v: lanes0-15 M=v, lanes16-31 M=v+8; N=lane&15
#pragma unroll
    for (int ti = 0; ti < 2; ++ti)
#pragma unroll
        for (int tj = 0; tj < 2; ++tj) {
            int n = nBlock + wn * 32 + tj * 16 + mr;
#pragma unroll
            for (int v = 0; v < 8; ++v) {
                int m = mBlock + wm * 32 + ti * 16 + v + 8 * half;
                Y[(size_t)m * N + n] = acc[ti][tj][v];
            }
        }
}

// ---------------- per-channel stats (deterministic two-stage) ----------------
__global__ __launch_bounds__(256) void stats_partial(
    const float* __restrict__ Y, int M,
    float* __restrict__ partS, float* __restrict__ partQ)
{
    int c = threadIdx.x;
    int blk = blockIdx.x;            // 64 blocks
    int rowsPer = M >> 6;
    float s = 0.f, q = 0.f;
    for (int r = blk * rowsPer; r < (blk + 1) * rowsPer; ++r) {
        float v = Y[(size_t)r * COUT + c];
        s += v; q += v * v;
    }
    partS[blk * COUT + c] = s;
    partQ[blk * COUT + c] = q;
}

__global__ __launch_bounds__(256) void stats_reduce(
    const float* __restrict__ partS, const float* __restrict__ partQ,
    float invM, float* __restrict__ meanOut, float* __restrict__ rstdOut)
{
    int c = threadIdx.x;
    float s = 0.f, q = 0.f;
    for (int b = 0; b < 64; ++b) { s += partS[b * COUT + c]; q += partQ[b * COUT + c]; }
    float mean = s * invM;
    float var  = q * invM - mean * mean;
    meanOut[c] = mean;
    rstdOut[c] = rsqrtf(var + 1e-5f);
}

__global__ __launch_bounds__(256) void bn_relu_inplace(
    float* __restrict__ Y, const float* __restrict__ mean,
    const float* __restrict__ rstd, const float* __restrict__ gamma,
    const float* __restrict__ beta)
{
    size_t i = (size_t)blockIdx.x * 256 + threadIdx.x;
    int c = (int)(i & (COUT - 1));
    float v = (Y[i] - mean[c]) * rstd[c] * gamma[c] + beta[c];
    Y[i] = v > 0.f ? v : 0.f;
}

// ---------------- 3-NN with inverse-distance weights ----------------
__global__ __launch_bounds__(256) void knn3_kernel(
    const float* __restrict__ p1, const float* __restrict__ p2,
    int* __restrict__ idxOut, float* __restrict__ wOut)
{
    __shared__ float sx[NP2], sy[NP2], sz[NP2];   // 24 KB of 320 KB LDS
    const int b = blockIdx.y;
    const float* p2b = p2 + (size_t)b * NP2 * 3;
    for (int j = threadIdx.x; j < NP2; j += 256) {
        sx[j] = p2b[j * 3 + 0];
        sy[j] = p2b[j * 3 + 1];
        sz[j] = p2b[j * 3 + 2];
    }
    __syncthreads();

    const int n = blockIdx.x * 256 + threadIdx.x;
    const float* p = p1 + ((size_t)b * NP1 + n) * 3;
    const float px = p[0], py = p[1], pz = p[2];

    float d0 = 3.4e38f, d1 = 3.4e38f, d2 = 3.4e38f;
    int i0 = 0, i1 = 0, i2 = 0;
    for (int j = 0; j < NP2; ++j) {
        float dx = px - sx[j], dy = py - sy[j], dz = pz - sz[j];
        float d = dx * dx + dy * dy + dz * dz;
        if (d < d0)      { d2 = d1; i2 = i1; d1 = d0; i1 = i0; d0 = d; i0 = j; }
        else if (d < d1) { d2 = d1; i2 = i1; d1 = d;  i1 = j; }
        else if (d < d2) { d2 = d;  i2 = j; }
    }
    float r0 = 1.f / (d0 + 1e-8f), r1 = 1.f / (d1 + 1e-8f), r2 = 1.f / (d2 + 1e-8f);
    float rs = 1.f / (r0 + r1 + r2);
    size_t o = ((size_t)b * NP1 + n) * 3;
    idxOut[o] = i0; idxOut[o + 1] = i1; idxOut[o + 2] = i2;
    wOut[o] = r0 * rs; wOut[o + 1] = r1 * rs; wOut[o + 2] = r2 * rs;
}

// ---------------- fused finale: BN+ReLU(y1) + weighted 3-row gather of h2 -----
__global__ __launch_bounds__(256) void final_kernel(
    float* __restrict__ Y1,                 // d_out, staged x1@W1+b1, in-place
    const float* __restrict__ mean1, const float* __restrict__ rstd1,
    const float* __restrict__ gamma1, const float* __restrict__ beta1,
    const float* __restrict__ H2,           // [B*NP2, COUT] normalized
    const int* __restrict__ idx, const float* __restrict__ w)
{
    const int point = blockIdx.x;           // 0 .. B*NP1-1
    const int b = point >> 13;              // / NP1
    const int c = threadIdx.x;
    const size_t o3 = (size_t)point * 3;
    const int   j0 = idx[o3], j1 = idx[o3 + 1], j2 = idx[o3 + 2];
    const float w0 = w[o3],  w1 = w[o3 + 1],  w2 = w[o3 + 2];
    const float* h2b = H2 + (size_t)b * NP2 * COUT;
    float g = w0 * h2b[(size_t)j0 * COUT + c]
            + w1 * h2b[(size_t)j1 * COUT + c]
            + w2 * h2b[(size_t)j2 * COUT + c];
    const size_t oi = (size_t)point * COUT + c;
    float v = (Y1[oi] - mean1[c]) * rstd1[c] * gamma1[c] + beta1[c];
    v = v > 0.f ? v : 0.f;
    Y1[oi] = v + g;
}

// ---------------- launch ----------------
extern "C" void kernel_launch(void* const* d_in, const int* in_sizes, int n_in,
                              void* d_out, int out_size, void* d_ws, size_t ws_size,
                              hipStream_t stream) {
    const float* p1     = (const float*)d_in[0];
    const float* x1     = (const float*)d_in[1];
    const float* p2     = (const float*)d_in[2];
    const float* x2     = (const float*)d_in[3];
    const float* W1     = (const float*)d_in[4];
    const float* b1     = (const float*)d_in[5];
    const float* gamma1 = (const float*)d_in[6];
    const float* beta1  = (const float*)d_in[7];
    const float* W2     = (const float*)d_in[8];
    const float* b2     = (const float*)d_in[9];
    const float* gamma2 = (const float*)d_in[10];
    const float* beta2  = (const float*)d_in[11];
    float* out = (float*)d_out;

    const int M2 = BATCH * NP2;   // 8192
    const int M1 = BATCH * NP1;   // 32768

    // workspace carve-up
    char* ws = (char*)d_ws;
    float* y2    = (float*)ws;            ws += (size_t)M2 * COUT * 4;        // 8 MB
    unsigned short* x1b = (unsigned short*)ws; ws += (size_t)M1 * COUT * 2;   // 16 MB
    unsigned short* x2b = (unsigned short*)ws; ws += (size_t)M2 * CIN  * 2;   // 8 MB
    unsigned short* W1t = (unsigned short*)ws; ws += (size_t)COUT * COUT * 2; // 128 KB
    unsigned short* W2t = (unsigned short*)ws; ws += (size_t)COUT * CIN  * 2; // 256 KB
    float* partS = (float*)ws;            ws += 64 * COUT * 4;
    float* partQ = (float*)ws;            ws += 64 * COUT * 4;
    float* mean2 = (float*)ws;            ws += COUT * 4;
    float* rstd2 = (float*)ws;            ws += COUT * 4;
    float* mean1 = (float*)ws;            ws += COUT * 4;
    float* rstd1 = (float*)ws;            ws += COUT * 4;
    int*   knnI  = (int*)ws;              ws += (size_t)M1 * 3 * 4;
    float* knnW  = (float*)ws;

    // 0) one-shot precision/layout conversions
    cvt_bf16_kernel<<<(M1 * COUT / 8 + 255) / 256, 256, 0, stream>>>(x1, x1b, M1 * COUT / 8);
    cvt_bf16_kernel<<<(M2 * CIN  / 8 + 255) / 256, 256, 0, stream>>>(x2, x2b, M2 * CIN / 8);
    cvt_w_transpose_kernel<<<COUT, 256, 0, stream>>>(W1, W1t, COUT);  // K=256
    cvt_w_transpose_kernel<<<CIN,  256, 0, stream>>>(W2, W2t, CIN);   // K=512
    // 1) y2 = x2 @ W2 + b2
    gemm_bias_bf16<<<dim3(COUT / BN, M2 / BM), 256, 0, stream>>>(
        x2b, W2t, b2, y2, M2, COUT, CIN);
    // 2-3) channel stats of y2
    stats_partial<<<64, 256, 0, stream>>>(y2, M2, partS, partQ);
    stats_reduce<<<1, 256, 0, stream>>>(partS, partQ, 1.f / (float)M2, mean2, rstd2);
    // 4) h2 = relu(BN(y2)) in place
    bn_relu_inplace<<<(M2 * COUT) / 256, 256, 0, stream>>>(y2, mean2, rstd2, gamma2, beta2);
    // 5) 3-NN + weights
    knn3_kernel<<<dim3(NP1 / 256, BATCH), 256, 0, stream>>>(p1, p2, knnI, knnW);
    // 6) y1 = x1 @ W1 + b1  (staged into d_out)
    gemm_bias_bf16<<<dim3(COUT / BN, M1 / BM), 256, 0, stream>>>(
        x1b, W1t, b1, out, M1, COUT, COUT);
    // 7-8) channel stats of y1
    stats_partial<<<64, 256, 0, stream>>>(out, M1, partS, partQ);
    stats_reduce<<<1, 256, 0, stream>>>(partS, partQ, 1.f / (float)M1, mean1, rstd1);
    // 9) out = relu(BN(y1)) + interpolate(h2)
    final_kernel<<<M1, 256, 0, stream>>>(out, mean1, rstd1, gamma1, beta1,
                                         y2, knnI, knnW);
}
